// ContextualAttention_4260607558172
// MI455X (gfx1250) — matrix operand order repaired and checked by reading, not compiled, not run on hardware
//
#include <hip/hip_runtime.h>
#include <hip/hip_bf16.h>
#include <math.h>

#define HDIM  1024
#define BATCH 8
#define SEQ   2048

typedef __bf16 bf16_t;
typedef __attribute__((ext_vector_type(16))) __bf16 v16bf;
typedef __attribute__((ext_vector_type(8)))  float  v8f;
typedef int v4i __attribute__((vector_size(16)));   // matches builtin's pointee type

union FragB16 { v16bf v; uint4 u[2]; };
union Acc8    { v8f  v; float f[8]; };

#define GLOBAL_AS __attribute__((address_space(1)))
#define LDS_AS    __attribute__((address_space(3)))

#if defined(__has_builtin)
#if __has_builtin(__builtin_amdgcn_global_load_async_to_lds_b128) && \
    __has_builtin(__builtin_amdgcn_s_wait_asynccnt)
#define HAVE_ASYNC_LDS 1
#endif
#endif

// one per-lane 16-byte async (direct-to-LDS) copy; sync fallback otherwise
__device__ __forceinline__ void copy_b128_to_lds(const bf16_t* g, bf16_t* l) {
#if defined(HAVE_ASYNC_LDS)
    const GLOBAL_AS void* gv = (const GLOBAL_AS void*)g;
    LDS_AS void*          lv = (LDS_AS void*)l;
    __builtin_amdgcn_global_load_async_to_lds_b128(
        (GLOBAL_AS v4i*)gv, (LDS_AS v4i*)lv, 0, 0);
#else
    *(uint4*)l = *(const uint4*)g;
#endif
}

// ---------------------------------------------------------------------------
// Shared epilogue:
// EPI: 0 = f32 out + bias              (Q/K projection)
//      1 = bf16 transposed out + bias  (V -> Vt[b][h][s])
//      2 = f32 out * scale             (scores)
//      3 = f32 out                     (attention output)
// C/D layout: lane holds col (lane&15), row = (lane>>4)*8 + r
// ---------------------------------------------------------------------------
template <int EPI>
__device__ __forceinline__ void store_epilogue(
    Acc8 (&acc)[4][2], float* __restrict__ Cptr, bf16_t* __restrict__ Cbf,
    const float* __restrict__ bias, int ldc, int z, long long sC, float scale,
    long long rowbase, long long colbase, int wm, int wn, int lrow, int lhi)
{
#pragma unroll
    for (int i = 0; i < 4; ++i) {
#pragma unroll
        for (int j = 0; j < 2; ++j) {
            const long long c  = colbase + wn * 32 + j * 16 + lrow;
            const long long r0 = rowbase + wm * 64 + i * 16 + lhi * 8;
            float bv = 0.0f;
            if (EPI == 0 || EPI == 1) bv = bias[c];
#pragma unroll
            for (int r = 0; r < 8; ++r) {
                const long long row = r0 + r;
                const float val = acc[i][j].f[r];
                if (EPI == 0) {
                    Cptr[(long long)z * sC + row * ldc + c] = val + bv;
                } else if (EPI == 1) {
                    const long long bb = row >> 11;        // / SEQ
                    const long long ss = row & (SEQ - 1);
                    Cbf[bb * (long long)HDIM * SEQ + c * SEQ + ss] = (bf16_t)(val + bv);
                } else if (EPI == 2) {
                    Cptr[(long long)z * sC + row * ldc + c] = val * scale;
                } else {
                    Cptr[(long long)z * sC + row * ldc + c] = val;
                }
            }
        }
    }
}

// ---------------------------------------------------------------------------
// C[M,N] = A[M,K] * B[N,K]^T, f32 sources converted to bf16 while staging.
// Used for QKV projections (hidden/W are f32 in HBM).
// ---------------------------------------------------------------------------
template <int EPI>
__global__ __launch_bounds__(256) void gemm_wmma_f32in(
    const float* __restrict__ Af, const float* __restrict__ Bf,
    float* __restrict__ Cptr, bf16_t* __restrict__ Cbf,
    const float* __restrict__ bias,
    int lda, int ldb, int ldc, int Kdim, float scale)
{
    constexpr int BM = 128, BN = 128, BK = 32;
    constexpr int LSTR = 40; // 32 + 8 halves pad (80B rows, 16B aligned)
    __shared__ bf16_t lA[BM * LSTR];
    __shared__ bf16_t lB[BN * LSTR];

    const int tid  = threadIdx.x;
    const int lane = tid & 31;
    const int wid  = tid >> 5;
    const int wm   = wid >> 2;     // 0..1  (64-row slab)
    const int wn   = wid & 3;      // 0..3  (32-col slab)
    const int lrow = lane & 15;
    const int lhi  = lane >> 4;    // 0/1

    const long long rowbase = (long long)blockIdx.y * BM;
    const long long colbase = (long long)blockIdx.x * BN;

    Acc8 acc[4][2];
#pragma unroll
    for (int i = 0; i < 4; ++i)
#pragma unroll
        for (int j = 0; j < 2; ++j)
#pragma unroll
            for (int r = 0; r < 8; ++r) acc[i][j].f[r] = 0.0f;

    const int lr = tid >> 1;        // 0..127 : tile row this thread loads
    const int lc = (tid & 1) << 4;  // 0 / 16 : 16 halves per thread

    for (int k0 = 0; k0 < Kdim; k0 += BK) {
        {   // stage A tile, f32 -> bf16
            bf16_t* dst = &lA[lr * LSTR + lc];
            const float* p = Af + (rowbase + lr) * (long long)lda + k0 + lc;
            union { uint4 u[2]; bf16_t h[16]; } pk;
#pragma unroll
            for (int e = 0; e < 4; ++e) {
                float4 x = ((const float4*)p)[e];
                pk.h[4*e+0] = (bf16_t)x.x; pk.h[4*e+1] = (bf16_t)x.y;
                pk.h[4*e+2] = (bf16_t)x.z; pk.h[4*e+3] = (bf16_t)x.w;
            }
            ((uint4*)dst)[0] = pk.u[0];
            *(uint4*)(dst + 8) = pk.u[1];
            if (k0 + BK < Kdim) __builtin_prefetch(p + BK, 0, 0);
        }
        {   // stage B tile, f32 -> bf16
            bf16_t* dst = &lB[lr * LSTR + lc];
            const float* p = Bf + (colbase + lr) * (long long)ldb + k0 + lc;
            union { uint4 u[2]; bf16_t h[16]; } pk;
#pragma unroll
            for (int e = 0; e < 4; ++e) {
                float4 x = ((const float4*)p)[e];
                pk.h[4*e+0] = (bf16_t)x.x; pk.h[4*e+1] = (bf16_t)x.y;
                pk.h[4*e+2] = (bf16_t)x.z; pk.h[4*e+3] = (bf16_t)x.w;
            }
            ((uint4*)dst)[0] = pk.u[0];
            *(uint4*)(dst + 8) = pk.u[1];
            if (k0 + BK < Kdim) __builtin_prefetch(p + BK, 0, 0);
        }
        __syncthreads();

        FragB16 fa[4], fb[2];
#pragma unroll
        for (int i = 0; i < 4; ++i) {
            // A: lane holds row (lane&15), k = (e>>3)*16 + (lane>>4)*8 + (e&7)
            const bf16_t* s = &lA[(wm * 64 + i * 16 + lrow) * LSTR + lhi * 8];
            fa[i].u[0] = *(const uint4*)s;
            fa[i].u[1] = *(const uint4*)(s + 16);
        }
#pragma unroll
        for (int j = 0; j < 2; ++j) {
            // B: lane holds col (lane&15), k = (e>>3)*16 + (lane>>4)*8 + (e&7)
            const bf16_t* s = &lB[(wn * 32 + j * 16 + lrow) * LSTR + lhi * 8];
            fb[j].u[0] = *(const uint4*)s;
            fb[j].u[1] = *(const uint4*)(s + 16);
        }
#pragma unroll
        for (int i = 0; i < 4; ++i)
#pragma unroll
            for (int j = 0; j < 2; ++j)
                acc[i][j].v = __builtin_amdgcn_wmma_f32_16x16x32_bf16(
                    false, fa[i].v, false, fb[j].v,
                    (short)0, acc[i][j].v, false, false);
        __syncthreads();
    }

    store_epilogue<EPI>(acc, Cptr, Cbf, bias, ldc, 0, 0, scale,
                        rowbase, colbase, wm, wn, lrow, lhi);
}

// ---------------------------------------------------------------------------
// bf16 x bf16 GEMM with double-buffered async global->LDS staging (no VGPR
// round trip; ASYNCcnt-pipelined). Used for scores and probs*V.
// ---------------------------------------------------------------------------
template <int EPI>
__global__ __launch_bounds__(256) void gemm_wmma_async(
    const bf16_t* __restrict__ Abase, const bf16_t* __restrict__ Bbase,
    float* __restrict__ Cptr,
    int lda, int ldb, int ldc, int Kdim,
    long long sA, long long sB, long long sC, float scale)
{
    constexpr int BM = 128, BN = 128, BK = 32;
    constexpr int LSTR = 40;
    constexpr int TILE = BM * LSTR;
    __shared__ bf16_t lA[2 * TILE];
    __shared__ bf16_t lB[2 * TILE];

    const int tid  = threadIdx.x;
    const int lane = tid & 31;
    const int wid  = tid >> 5;
    const int wm   = wid >> 2;
    const int wn   = wid & 3;
    const int lrow = lane & 15;
    const int lhi  = lane >> 4;

    const int z = blockIdx.z;
    const long long rowbase = (long long)blockIdx.y * BM;
    const long long colbase = (long long)blockIdx.x * BN;

    const bf16_t* Ah = Abase + (long long)z * sA;
    const bf16_t* Bh = Bbase + (long long)z * sB;

    Acc8 acc[4][2];
#pragma unroll
    for (int i = 0; i < 4; ++i)
#pragma unroll
        for (int j = 0; j < 2; ++j)
#pragma unroll
            for (int r = 0; r < 8; ++r) acc[i][j].f[r] = 0.0f;

    const int lr = tid >> 1;
    const int lc = (tid & 1) << 4;

    const bf16_t* gA = Ah + (rowbase + lr) * (long long)lda + lc;
    const bf16_t* gB = Bh + (colbase + lr) * (long long)ldb + lc;

    // Issue the 4 x b128 transfers that stage one K-tile into buffer `buf`.
    auto issue_tile = [&](int k0, int buf) {
        bf16_t* dA = &lA[buf * TILE + lr * LSTR + lc];
        bf16_t* dB = &lB[buf * TILE + lr * LSTR + lc];
        copy_b128_to_lds(gA + k0,     dA);
        copy_b128_to_lds(gA + k0 + 8, dA + 8);
        copy_b128_to_lds(gB + k0,     dB);
        copy_b128_to_lds(gB + k0 + 8, dB + 8);
    };

    issue_tile(0, 0);   // prologue: stage tile 0

    const int nk = Kdim / BK;
    for (int kt = 0; kt < nk; ++kt) {
        const int cur = kt & 1;
        if (kt + 1 < nk) {
            issue_tile((kt + 1) * BK, cur ^ 1);   // overlap next tile's copy
#if defined(HAVE_ASYNC_LDS)
            // async loads complete in order: <=4 outstanding => tile `kt` landed
            __builtin_amdgcn_s_wait_asynccnt(4);
#endif
        } else {
#if defined(HAVE_ASYNC_LDS)
            __builtin_amdgcn_s_wait_asynccnt(0);
#endif
        }
        __syncthreads();

        const bf16_t* bA = &lA[cur * TILE];
        const bf16_t* bB = &lB[cur * TILE];
        FragB16 fa[4], fb[2];
#pragma unroll
        for (int i = 0; i < 4; ++i) {
            const bf16_t* s = &bA[(wm * 64 + i * 16 + lrow) * LSTR + lhi * 8];
            fa[i].u[0] = *(const uint4*)s;
            fa[i].u[1] = *(const uint4*)(s + 16);
        }
#pragma unroll
        for (int j = 0; j < 2; ++j) {
            const bf16_t* s = &bB[(wn * 32 + j * 16 + lrow) * LSTR + lhi * 8];
            fb[j].u[0] = *(const uint4*)s;
            fb[j].u[1] = *(const uint4*)(s + 16);
        }
#pragma unroll
        for (int i = 0; i < 4; ++i)
#pragma unroll
            for (int j = 0; j < 2; ++j)
                acc[i][j].v = __builtin_amdgcn_wmma_f32_16x16x32_bf16(
                    false, fa[i].v, false, fb[j].v,
                    (short)0, acc[i][j].v, false, false);
        __syncthreads();   // all reads of buf `cur` done before it is re-filled
    }

    store_epilogue<EPI>(acc, Cptr, nullptr, nullptr, ldc, z, sC, scale,
                        rowbase, colbase, wm, wn, lrow, lhi);
}

// ---------------------------------------------------------------------------
// Per-batch context gate scalars: cq[b] = ctx[b]·g_qc, ck[b] = ctx[b]·g_kc
// ---------------------------------------------------------------------------
__global__ __launch_bounds__(256) void ctx_gate_kernel(
    const float* __restrict__ ctx, const float* __restrict__ gqc,
    const float* __restrict__ gkc, float* __restrict__ cq, float* __restrict__ ck)
{
    const int b = blockIdx.x, tid = threadIdx.x;
    const int h = tid * 4;
    float4 c = *(const float4*)(ctx + b * HDIM + h);
    float4 q = *(const float4*)(gqc + h);
    float4 k = *(const float4*)(gkc + h);
    float pq = c.x*q.x + c.y*q.y + c.z*q.z + c.w*q.w;
    float pk = c.x*k.x + c.y*k.y + c.z*k.z + c.w*k.w;
    __shared__ float red[256];
    red[tid] = pq; __syncthreads();
    for (int s = 128; s > 0; s >>= 1) { if (tid < s) red[tid] += red[tid + s]; __syncthreads(); }
    if (tid == 0) cq[b] = red[0];
    __syncthreads();
    red[tid] = pk; __syncthreads();
    for (int s = 128; s > 0; s >>= 1) { if (tid < s) red[tid] += red[tid + s]; __syncthreads(); }
    if (tid == 0) ck[b] = red[0];
}

// ---------------------------------------------------------------------------
// ctxq[b][h] = ctx[b]·Wcq[h,:], ctxk[b][h] = ctx[b]·Wck[h,:]
// ---------------------------------------------------------------------------
__global__ __launch_bounds__(256) void ctx_proj_kernel(
    const float* __restrict__ ctx, const float* __restrict__ Wcq,
    const float* __restrict__ Wck, float* __restrict__ ctxq, float* __restrict__ ctxk)
{
    const int h = blockIdx.x * 256 + threadIdx.x;
    const int b = blockIdx.y;
    const float* cv = ctx + b * HDIM;
    const float* wq = Wcq + (long long)h * HDIM;
    const float* wk = Wck + (long long)h * HDIM;
    float sq = 0.f, sk = 0.f;
    for (int i = 0; i < HDIM; i += 4) {
        float4 c = *(const float4*)(cv + i);
        float4 a = *(const float4*)(wq + i);
        float4 d = *(const float4*)(wk + i);
        sq += c.x*a.x + c.y*a.y + c.z*a.z + c.w*a.w;
        sk += c.x*d.x + c.y*d.y + c.z*d.z + c.w*d.w;
    }
    ctxq[b * HDIM + h] = sq;
    ctxk[b * HDIM + h] = sk;
}

// ---------------------------------------------------------------------------
// Gate + blend, in place: f32 row [1024] -> bf16 row [1024] at same row base
// (row byte stride stays 4096 => bf16 lda = 2048 halves)
// ---------------------------------------------------------------------------
__global__ __launch_bounds__(256) void gate_blend_kernel(
    float* __restrict__ QK, const float* __restrict__ gh,
    const float* __restrict__ cgate, const float* __restrict__ ctxp)
{
    const long long row = blockIdx.x;
    const int b = (int)(row >> 11);
    const int tid = threadIdx.x;
    const int h = tid * 4;
    float4 q = *(const float4*)(QK + row * HDIM + h);
    float4 g = *(const float4*)(gh + h);
    float part = q.x*g.x + q.y*g.y + q.z*g.z + q.w*g.w;
    __shared__ float red[256];
    red[tid] = part; __syncthreads();
    for (int s = 128; s > 0; s >>= 1) { if (tid < s) red[tid] += red[tid + s]; __syncthreads(); }
    const float gate = 1.0f / (1.0f + __expf(-(red[0] + cgate[b])));
    float4 c = *(const float4*)(ctxp + b * HDIM + h);
    union { uint2 u; bf16_t h4[4]; } pk;
    pk.h4[0] = (bf16_t)((1.0f - gate) * q.x + gate * c.x);
    pk.h4[1] = (bf16_t)((1.0f - gate) * q.y + gate * c.y);
    pk.h4[2] = (bf16_t)((1.0f - gate) * q.z + gate * c.z);
    pk.h4[3] = (bf16_t)((1.0f - gate) * q.w + gate * c.w);
    bf16_t* dst = (bf16_t*)QK + row * 2048 + h;   // same row byte offset, half width
    *(uint2*)dst = pk.u;
}

// ---------------------------------------------------------------------------
// Row softmax: scores f32 [row][2048] -> probs bf16 [row][2048]
// ---------------------------------------------------------------------------
__global__ __launch_bounds__(256) void softmax_kernel(
    const float* __restrict__ S, bf16_t* __restrict__ P)
{
    const long long row = blockIdx.x;
    const int tid = threadIdx.x;
    const float* sr = S + row * SEQ;
    float v[8];
    float4 a = *(const float4*)(sr + tid * 8);
    float4 b = *(const float4*)(sr + tid * 8 + 4);
    v[0]=a.x; v[1]=a.y; v[2]=a.z; v[3]=a.w; v[4]=b.x; v[5]=b.y; v[6]=b.z; v[7]=b.w;
    float m = v[0];
#pragma unroll
    for (int i = 1; i < 8; ++i) m = fmaxf(m, v[i]);
    __shared__ float red[256];
    red[tid] = m; __syncthreads();
    for (int s = 128; s > 0; s >>= 1) { if (tid < s) red[tid] = fmaxf(red[tid], red[tid + s]); __syncthreads(); }
    m = red[0]; __syncthreads();
    float sum = 0.f;
#pragma unroll
    for (int i = 0; i < 8; ++i) { v[i] = __expf(v[i] - m); sum += v[i]; }
    red[tid] = sum; __syncthreads();
    for (int s = 128; s > 0; s >>= 1) { if (tid < s) red[tid] += red[tid + s]; __syncthreads(); }
    const float inv = 1.0f / red[0];
    union { uint4 u; bf16_t h[8]; } pk;
#pragma unroll
    for (int i = 0; i < 8; ++i) pk.h[i] = (bf16_t)(v[i] * inv);
    *(uint4*)(P + row * SEQ + tid * 8) = pk.u;
}

// ---------------------------------------------------------------------------
extern "C" void kernel_launch(void* const* d_in, const int* in_sizes, int n_in,
                              void* d_out, int out_size, void* d_ws, size_t ws_size,
                              hipStream_t stream) {
    const float* hidden = (const float*)d_in[0];
    const float* ctx    = (const float*)d_in[1];
    const float* Wq  = (const float*)d_in[2];  const float* bq = (const float*)d_in[3];
    const float* Wk  = (const float*)d_in[4];  const float* bk = (const float*)d_in[5];
    const float* Wv  = (const float*)d_in[6];  const float* bv = (const float*)d_in[7];
    const float* Wcq = (const float*)d_in[8];  const float* Wck = (const float*)d_in[9];
    const float* g_qh = (const float*)d_in[10]; const float* g_kh = (const float*)d_in[11];
    const float* g_qc = (const float*)d_in[12]; const float* g_kc = (const float*)d_in[13];

    float* out    = (float*)d_out;                               // [B,S,H]
    float* scores = out + (long long)BATCH * SEQ * HDIM;         // [B,S,S]

    char* w = (char*)d_ws;
    float* Qf = (float*)w;  w += (long long)BATCH * SEQ * HDIM * 4;   // 64MB (later: Qhat bf16 strided, then probs bf16)
    float* Kf = (float*)w;  w += (long long)BATCH * SEQ * HDIM * 4;   // 64MB (later: Khat bf16 strided)
    bf16_t* Vt = (bf16_t*)w; w += (long long)BATCH * HDIM * SEQ * 2;  // 32MB  Vt[b][h][s]
    float* ctxq = (float*)w; w += BATCH * HDIM * 4;
    float* ctxk = (float*)w; w += BATCH * HDIM * 4;
    float* cq   = (float*)w; w += BATCH * 4;
    float* ck   = (float*)w; w += BATCH * 4;

    const dim3 blk(256);

    // 1) QKV projections: [16384,1024] x [1024,1024]^T  (f32 in, bf16 compute)
    const dim3 gproj(HDIM / 128, (BATCH * SEQ) / 128, 1);
    gemm_wmma_f32in<0><<<gproj, blk, 0, stream>>>(hidden, Wq, Qf, nullptr, bq,
        HDIM, HDIM, HDIM, HDIM, 1.0f);
    gemm_wmma_f32in<0><<<gproj, blk, 0, stream>>>(hidden, Wk, Kf, nullptr, bk,
        HDIM, HDIM, HDIM, HDIM, 1.0f);
    gemm_wmma_f32in<1><<<gproj, blk, 0, stream>>>(hidden, Wv, nullptr, Vt, bv,
        HDIM, HDIM, HDIM, HDIM, 1.0f);

    // 2) context scalars + projections
    ctx_gate_kernel<<<BATCH, blk, 0, stream>>>(ctx, g_qc, g_kc, cq, ck);
    ctx_proj_kernel<<<dim3(HDIM / 256, BATCH), blk, 0, stream>>>(ctx, Wcq, Wck, ctxq, ctxk);

    // 3) gate + blend, in-place f32 -> bf16 (Qhat/Khat, lda = 2048 halves)
    gate_blend_kernel<<<BATCH * SEQ, blk, 0, stream>>>(Qf, g_qh, cq, ctxq);
    gate_blend_kernel<<<BATCH * SEQ, blk, 0, stream>>>(Kf, g_kh, ck, ctxk);

    // 4) scores = Qhat·Khat^T * 1/sqrt(H)  -> d_out scores region (f32)
    //    bf16 operands: async double-buffered global->LDS pipeline
    const dim3 gscore(SEQ / 128, SEQ / 128, BATCH);
    gemm_wmma_async<2><<<gscore, blk, 0, stream>>>((bf16_t*)Qf, (bf16_t*)Kf, scores,
        2048, 2048, SEQ, HDIM,
        (long long)SEQ * 2048, (long long)SEQ * 2048, (long long)SEQ * SEQ, 0.03125f);

    // 5) softmax -> probs bf16, aliased over Qf (Qhat dead after scores GEMM)
    bf16_t* probs = (bf16_t*)Qf;
    softmax_kernel<<<BATCH * SEQ, blk, 0, stream>>>(scores, probs);

    // 6) output = probs · V  (B = Vt[N=H][K=S])
    const dim3 gout(HDIM / 128, SEQ / 128, BATCH);
    gemm_wmma_async<3><<<gout, blk, 0, stream>>>(probs, Vt, out,
        2048, 2048, HDIM, SEQ,
        (long long)SEQ * 2048, (long long)HDIM * SEQ, (long long)SEQ * HDIM, 1.0f);
}
